// MyMultiHeadAttention_60567628808515
// MI455X (gfx1250) — compile-verified, matrix operand order checked by reading, not run on hardware
//
#include <hip/hip_runtime.h>

#define D_MODEL 1024
#define NHEAD   16
#define HEAD_DIM 64
#define BB      8
#define SS      1024
#define KNEG    (-1e9f)

typedef __attribute__((ext_vector_type(16))) __bf16 v16bf;
typedef __attribute__((ext_vector_type(8)))  __bf16 v8bf;
typedef __attribute__((ext_vector_type(8)))  float  v8f;
typedef __attribute__((ext_vector_type(4)))  float  v4f;

union V16 { v16bf v; v8bf h[2]; };

// D = A(16x32 bf16) * B(32x16 bf16) + C(16x16 f32)   [v_wmma_f32_16x16x32_bf16]
__device__ __forceinline__ v8f wmma_bf16(v16bf a, v16bf b, v8f c) {
  return __builtin_amdgcn_wmma_f32_16x16x32_bf16(false, a, false, b, (short)0, c,
                                                 false, false);
}

// ---- CDNA5 async global->LDS copy (ASYNCcnt path), inline asm (portable) ----
// the vdst VGPR supplies the LDS byte address; generic shared pointers carry the workgroup
// LDS offset in their low 32 bits (aperture form), so truncation is the offset.
__device__ __forceinline__ unsigned lds_off(const void* p) {
  return (unsigned)(uintptr_t)p;
}
__device__ __forceinline__ void async_copy_b128(unsigned ldsoff, const void* g) {
  asm volatile("global_load_async_to_lds_b128 %0, %1, off"
               :: "v"(ldsoff), "v"(g) : "memory");
}
__device__ __forceinline__ void wait_async0() {
  asm volatile("s_wait_asynccnt 0x0" ::: "memory");
}

// Load a 16x32 (rows x K) operand tile from row-major fp32 (global), cvt bf16.
// ISA 7.12.2 16-bit layout: lanes 0-15 hold K in {0..7,16..23}, lanes 16-31
// hold K in {8..15,24..31}; row = lane & 15.
__device__ __forceinline__ v16bf load_tile_f32(const float* __restrict__ base,
                                               int ld, int r0, int kk, int lane) {
  const int half = lane >> 4;
  const int r    = lane & 15;
  const float* p = base + (size_t)(r0 + r) * ld + kk + 8 * half;
  V16 u;
#pragma unroll
  for (int i = 0; i < 8; ++i) {
    u.v[i]     = (__bf16)p[i];
    u.v[8 + i] = (__bf16)p[16 + i];
  }
  return u.v;
}

// Same layout, source already bf16 (two 16B vector loads).
__device__ __forceinline__ v16bf load_tile_bf16(const __bf16* __restrict__ base,
                                                int ld, int r0, int kk, int lane) {
  const int half = lane >> 4;
  const int r    = lane & 15;
  const __bf16* p = base + (size_t)(r0 + r) * ld + kk + 8 * half;
  V16 u;
  u.h[0] = *(const v8bf*)(p);
  u.h[1] = *(const v8bf*)(p + 16);
  return u.v;
}

// ---------------------------------------------------------------------------
// Kernel 1: fused Q/K/V projections, LDS-blocked GEMM.
// Workgroup (256 thr, 8 waves) owns a 128x128 C tile; wave wv owns rows
// [row0+16*wv, +16) x 128 cols = 8 independent 16x16 WMMA accumulators.
// A strips staged fp32 via GLOBAL_LOAD_ASYNC_TO_LDS_B128 (double-buffered);
// W tiles staged bf16 via regular loads + v_cvt_pk_bf16_f32 + ds_store.
//   qp[b,h,s,d] = (q @ wq.T + bq)/8   kp[b,h,s,d] = k @ wk.T
//   vt[b,h,d,s] =  v @ wv.T + bv
// ---------------------------------------------------------------------------
__global__ __launch_bounds__(256) void qkv_proj_kernel(
    const float* __restrict__ q, const float* __restrict__ k,
    const float* __restrict__ v,
    const float* __restrict__ wq, const float* __restrict__ bq,
    const float* __restrict__ wk,
    const float* __restrict__ wv, const float* __restrict__ bv,
    __bf16* __restrict__ qp, __bf16* __restrict__ kp, __bf16* __restrict__ vt)
{
  __shared__ alignas(16) float  sA[2][8][16][32];   // 32 KB: per-wave A strips
  __shared__ alignas(16) __bf16 sB[2][128][32];     // 16 KB: W tile (bf16)

  const int tid  = threadIdx.x;
  const int lane = tid & 31;
  const int wvid = tid >> 5;
  const int half = lane >> 4;
  const int r    = lane & 15;

  const int perMat = (BB * SS / 128) * (D_MODEL / 128);   // 64*8 = 512 blocks
  const int mat = blockIdx.x / perMat;                    // 0:Q 1:K 2:V
  const int bt  = blockIdx.x % perMat;
  const int tm  = bt / (D_MODEL / 128);
  const int tn  = bt % (D_MODEL / 128);
  const int row0 = tm * 128, col0 = tn * 128;

  const float* Ap = (mat == 0) ? q  : (mat == 1) ? k  : v;
  const float* Wp = (mat == 0) ? wq : (mat == 1) ? wk : wv;

  v8f acc[8];
#pragma unroll
  for (int ns = 0; ns < 8; ++ns) acc[ns] = (v8f){};

  v4f breg[4];                                      // in-flight W chunk (16 f32)

  auto stageA = [&](int buf, int kk) {              // 2KB strip, 4 async b128
    const char* gb = (const char*)(Ap + (size_t)(row0 + wvid * 16) * D_MODEL + kk);
    const unsigned lb = lds_off(&sA[buf][wvid][0][0]);
#pragma unroll
    for (int i = 0; i < 4; ++i) {
      const unsigned o = (unsigned)(i * 512 + lane * 16);     // strip byte off
      async_copy_b128(lb + o, gb + (size_t)(o >> 7) * (D_MODEL * 4) + (o & 127u));
    }
  };
  auto loadB = [&](int kk) {                        // thread: 16 f32 of W row
    const float* src = Wp + (size_t)(col0 + (tid >> 1)) * D_MODEL + kk + (tid & 1) * 16;
    __builtin_prefetch(src + 32, 0, 0);
#pragma unroll
    for (int i = 0; i < 4; ++i) breg[i] = *(const v4f*)(src + i * 4);
  };
  auto storeB = [&](int buf) {                      // cvt + 32B ds_store
    V16 u;
    const float* f = (const float*)breg;
#pragma unroll
    for (int i = 0; i < 16; ++i) u.v[i] = (__bf16)f[i];
    __bf16* dst = &sB[buf][tid >> 1][(tid & 1) * 16];
    *(v8bf*)dst       = u.h[0];
    *(v8bf*)(dst + 8) = u.h[1];
  };
  auto compute = [&](int buf) {
    V16 ua;                                         // A operand: LDS f32 -> bf16
    const float* ap = &sA[buf][wvid][r][0];
#pragma unroll
    for (int i = 0; i < 8; ++i) {
      ua.v[i]     = (__bf16)ap[8 * half + i];
      ua.v[8 + i] = (__bf16)ap[16 + 8 * half + i];
    }
#pragma unroll
    for (int ns = 0; ns < 8; ++ns) {                // 8 WMMAs reuse one A fetch
      V16 ub;
      const __bf16* bp = &sB[buf][ns * 16 + r][0];
      ub.h[0] = *(const v8bf*)(bp + 8 * half);
      ub.h[1] = *(const v8bf*)(bp + 16 + 8 * half);
      acc[ns] = wmma_bf16(ua.v, ub.v, acc[ns]);
    }
  };

  stageA(0, 0);
  loadB(0);
  storeB(0);
  wait_async0();
  __syncthreads();

  const int NT = D_MODEL / 32;
  for (int t = 0; t < NT; ++t) {
    const int  cur  = t & 1;
    const bool more = (t + 1 < NT);
    if (more) { loadB((t + 1) * 32); stageA(1 ^ cur, (t + 1) * 32); }
    compute(cur);                                   // overlaps async + loads
    if (more) { storeB(1 ^ cur); wait_async0(); }
    __syncthreads();
  }

  const int mrow0 = row0 + wvid * 16;
#pragma unroll
  for (int ns = 0; ns < 8; ++ns) {
    const int n = col0 + ns * 16 + r;
    const int h = n >> 6;
    const int d = n & 63;
#pragma unroll
    for (int j = 0; j < 8; ++j) {
      const int m  = mrow0 + j + 8 * half;
      const int bi = m >> 10;
      const int si = m & (SS - 1);
      float val = acc[ns][j];
      if (mat == 0) {
        val = (val + bq[n]) * 0.125f;               // fold 1/sqrt(Dh) into Q
        qp[(((size_t)bi * NHEAD + h) * SS + si) * HEAD_DIM + d] = (__bf16)val;
      } else if (mat == 1) {
        kp[(((size_t)bi * NHEAD + h) * SS + si) * HEAD_DIM + d] = (__bf16)val;
      } else {
        val += bv[n];
        vt[(((size_t)bi * NHEAD + h) * HEAD_DIM + d) * SS + si] = (__bf16)val;
      }
    }
  }
}

// ---------------------------------------------------------------------------
// Kernel 2: attention for one (b, h, 16-row q tile) per workgroup (4 waves).
//   scores(16x1024) -> LDS, masks applied, softmax, then attn @ V via WMMA.
// ---------------------------------------------------------------------------
__global__ __launch_bounds__(128) void attn_kernel(
    const __bf16* __restrict__ qp, const __bf16* __restrict__ kp,
    const __bf16* __restrict__ vt,
    const unsigned char* __restrict__ pad, const float* __restrict__ causal,
    __bf16* __restrict__ ao)
{
  __shared__ float sS[16][SS];      // 64 KB score / probability tile
  __shared__ float sMax[16];
  __shared__ float sSum[16];
  __shared__ float sRed[16][8];

  const int tid  = threadIdx.x;
  const int lane = tid & 31;
  const int wvid = tid >> 5;        // wave id 0..3
  const int half = lane >> 4;
  const int r    = lane & 15;

  const int bh = blockIdx.x / (SS / 16);
  const int qt = blockIdx.x % (SS / 16);
  const int bi = bh / NHEAD;
  const int h  = bh % NHEAD;
  const int q0 = qt * 16;

  const __bf16* Qb = qp + ((size_t)bi * NHEAD + h) * SS * HEAD_DIM;
  const __bf16* Kb = kp + ((size_t)bi * NHEAD + h) * SS * HEAD_DIM;
  const __bf16* Vb = vt + ((size_t)bi * NHEAD + h) * HEAD_DIM * SS;

  // Q operand for the two K-steps over HEAD_DIM=64; reused for all k tiles.
  const v16bf aq0 = load_tile_bf16(Qb, HEAD_DIM, q0, 0, lane);
  const v16bf aq1 = load_tile_bf16(Qb, HEAD_DIM, q0, 32, lane);

  // -------- Phase 1: scores = (Q/sqrt(d)) @ K^T, masked, into LDS ----------
  for (int kt = wvid; kt < SS / 16; kt += 4) {
    const int k0 = kt * 16;
    v8f acc = {};
    acc = wmma_bf16(aq0, load_tile_bf16(Kb, HEAD_DIM, k0, 0, lane), acc);
    acc = wmma_bf16(aq1, load_tile_bf16(Kb, HEAD_DIM, k0, 32, lane), acc);

    const int  kc     = k0 + r;
    const bool padded = pad[(size_t)bi * SS + kc] != 0;
#pragma unroll
    for (int j = 0; j < 8; ++j) {
      const int qr = j + 8 * half;
      const int qg = q0 + qr;
      float val = acc[j];
      if (padded) val = KNEG;                       // where(padding, NEG, s)
      val += causal[(size_t)qg * SS + kc];          // + causal additive mask
      sS[qr][kc] = val;
    }
  }
  __syncthreads();

  // -------- Phase 2: row softmax stats; exp stored back in place ----------
  {
    const int rr = tid & 15;
    const int g  = tid >> 4;
    const int c0 = g * (SS / 8);
    float m = -3.4e38f;
    for (int c = 0; c < SS / 8; ++c) m = fmaxf(m, sS[rr][c0 + c]);
    sRed[rr][g] = m;
    __syncthreads();
    if (g == 0) {
      float mm = sRed[rr][0];
#pragma unroll
      for (int i = 1; i < 8; ++i) mm = fmaxf(mm, sRed[rr][i]);
      sMax[rr] = mm;
    }
    __syncthreads();
    const float mm = sMax[rr];
    float sum = 0.f;
    for (int c = 0; c < SS / 8; ++c) {
      const float e = __expf(sS[rr][c0 + c] - mm);
      sS[rr][c0 + c] = e;
      sum += e;
    }
    sRed[rr][g] = sum;
    __syncthreads();
    if (g == 0) {
      float ss = 0.f;
#pragma unroll
      for (int i = 0; i < 8; ++i) ss += sRed[rr][i];
      sSum[rr] = ss;
    }
    __syncthreads();
  }

  // -------- Phase 3: out = (exp scores) @ V, each wave owns a d-stripe -----
  const int d0 = wvid * 16;
  v8f acc = {};
  for (int k0 = 0; k0 < SS; k0 += 32) {
    V16 ua;
    const float* prow = &sS[r][0];
#pragma unroll
    for (int i = 0; i < 8; ++i) {
      ua.v[i]     = (__bf16)prow[k0 + 8 * half + i];
      ua.v[8 + i] = (__bf16)prow[k0 + 16 + 8 * half + i];
    }
    const v16bf bv_ = load_tile_bf16(Vb, SS, d0, k0, lane);
    acc = wmma_bf16(ua.v, bv_, acc);
  }
#pragma unroll
  for (int j = 0; j < 8; ++j) {
    const int qr  = j + 8 * half;
    const float val = acc[j] / sSum[qr];
    const int   n   = h * HEAD_DIM + d0 + r;
    ao[((size_t)bi * SS + q0 + qr) * D_MODEL + n] = (__bf16)val;
  }
}

// ---------------------------------------------------------------------------
// Kernel 3: out = ao @ wo.T + bo  (fp32 result), LDS-blocked like kernel 1.
// A (ao) is already bf16 -> pure async-to-LDS staging, zero conversion.
// ---------------------------------------------------------------------------
__global__ __launch_bounds__(256) void out_proj_kernel(
    const __bf16* __restrict__ ao, const float* __restrict__ wo,
    const float* __restrict__ bo, float* __restrict__ out)
{
  __shared__ alignas(16) __bf16 sA[2][8][16][32];   // 16 KB bf16 A strips
  __shared__ alignas(16) __bf16 sB[2][128][32];     // 16 KB bf16 W tile

  const int tid  = threadIdx.x;
  const int lane = tid & 31;
  const int wvid = tid >> 5;
  const int half = lane >> 4;
  const int r    = lane & 15;

  const int tm   = blockIdx.x / (D_MODEL / 128);
  const int tn   = blockIdx.x % (D_MODEL / 128);
  const int row0 = tm * 128, col0 = tn * 128;

  v8f acc[8];
#pragma unroll
  for (int ns = 0; ns < 8; ++ns) acc[ns] = (v8f){};

  v4f breg[4];

  auto stageA = [&](int buf, int kk) {              // 1KB strip, 2 async b128
    const char* gb = (const char*)(ao + (size_t)(row0 + wvid * 16) * D_MODEL + kk);
    const unsigned lb = lds_off(&sA[buf][wvid][0][0]);
#pragma unroll
    for (int i = 0; i < 2; ++i) {
      const unsigned o = (unsigned)(i * 512 + lane * 16);
      async_copy_b128(lb + o, gb + (size_t)(o >> 6) * (D_MODEL * 2) + (o & 63u));
    }
  };
  auto loadB = [&](int kk) {
    const float* src = wo + (size_t)(col0 + (tid >> 1)) * D_MODEL + kk + (tid & 1) * 16;
    __builtin_prefetch(src + 32, 0, 0);
#pragma unroll
    for (int i = 0; i < 4; ++i) breg[i] = *(const v4f*)(src + i * 4);
  };
  auto storeB = [&](int buf) {
    V16 u;
    const float* f = (const float*)breg;
#pragma unroll
    for (int i = 0; i < 16; ++i) u.v[i] = (__bf16)f[i];
    __bf16* dst = &sB[buf][tid >> 1][(tid & 1) * 16];
    *(v8bf*)dst       = u.h[0];
    *(v8bf*)(dst + 8) = u.h[1];
  };
  auto compute = [&](int buf) {
    V16 ua;                                         // A operand straight bf16
    const __bf16* ap = &sA[buf][wvid][r][0];
    ua.h[0] = *(const v8bf*)(ap + 8 * half);
    ua.h[1] = *(const v8bf*)(ap + 16 + 8 * half);
#pragma unroll
    for (int ns = 0; ns < 8; ++ns) {
      V16 ub;
      const __bf16* bp = &sB[buf][ns * 16 + r][0];
      ub.h[0] = *(const v8bf*)(bp + 8 * half);
      ub.h[1] = *(const v8bf*)(bp + 16 + 8 * half);
      acc[ns] = wmma_bf16(ua.v, ub.v, acc[ns]);
    }
  };

  stageA(0, 0);
  loadB(0);
  storeB(0);
  wait_async0();
  __syncthreads();

  const int NT = D_MODEL / 32;
  for (int t = 0; t < NT; ++t) {
    const int  cur  = t & 1;
    const bool more = (t + 1 < NT);
    if (more) { loadB((t + 1) * 32); stageA(1 ^ cur, (t + 1) * 32); }
    compute(cur);
    if (more) { storeB(1 ^ cur); wait_async0(); }
    __syncthreads();
  }

  const int mrow0 = row0 + wvid * 16;
#pragma unroll
  for (int ns = 0; ns < 8; ++ns) {
    const int n = col0 + ns * 16 + r;
    const float bias = bo[n];
#pragma unroll
    for (int j = 0; j < 8; ++j) {
      const int m = mrow0 + j + 8 * half;
      out[(size_t)m * D_MODEL + n] = acc[ns][j] + bias;
    }
  }
}

extern "C" void kernel_launch(void* const* d_in, const int* in_sizes, int n_in,
                              void* d_out, int out_size, void* d_ws, size_t ws_size,
                              hipStream_t stream)
{
  const float*         q      = (const float*)d_in[0];
  const float*         k      = (const float*)d_in[1];
  const float*         v      = (const float*)d_in[2];
  const unsigned char* pad    = (const unsigned char*)d_in[3];
  const float*         causal = (const float*)d_in[4];
  const float*         wq     = (const float*)d_in[5];
  const float*         bq     = (const float*)d_in[6];
  const float*         wk     = (const float*)d_in[7];
  const float*         wv     = (const float*)d_in[8];
  const float*         bv     = (const float*)d_in[9];
  const float*         wo     = (const float*)d_in[10];
  const float*         bo     = (const float*)d_in[11];
  float*               out    = (float*)d_out;

  // Workspace: 4 bf16 intermediates of B*H*S*Dh = 8Mi elements (16 MB) each.
  const size_t HS = (size_t)BB * NHEAD * SS * HEAD_DIM;
  __bf16* qp = (__bf16*)d_ws;
  __bf16* kp = qp + HS;
  __bf16* vt = kp + HS;
  __bf16* ao = vt + HS;

  const int perMat = (BB * SS / 128) * (D_MODEL / 128);   // 512 blocks / GEMM

  qkv_proj_kernel<<<3 * perMat, 256, 0, stream>>>(q, k, v, wq, bq, wk, wv, bv,
                                                  qp, kp, vt);
  attn_kernel<<<BB * NHEAD * (SS / 16), 128, 0, stream>>>(qp, kp, vt, pad,
                                                          causal, ao);
  out_proj_kernel<<<perMat, 256, 0, stream>>>(ao, wo, bo, out);
}
